// Streamed_30700426232146
// MI455X (gfx1250) — matrix operand order repaired and checked by reading, not compiled, not run on hardware
//
#include <hip/hip_runtime.h>

// ---------------------------------------------------------------------------
// MoE linear (T=16384, 8 experts, 1024x1024) + bias + ReLU on gfx1250.
// bin tokens -> bf16 x rows -> bf16 WMMA-swizzled W -> bf16 WMMA GEMM with
// async global->LDS staging of B (triple buffered), f32 accumulate.
// Block: 8 waves = 2(M) x 4(N); wave tile M=32 x N=64; 8 v_wmma / 32-K chunk.
// ---------------------------------------------------------------------------

#define T_TOK   16384
#define DIN     1024
#define DOUT    1024
#define NEXP    8
#define KCH     32                  // K per v_wmma_f32_16x16x32_bf16
#define NCHUNKS (DIN / KCH)         // 32
#define BLOCK_N 256                 // N per block
#define WAVE_N  64                  // N per wave (4 x 16 tiles)
#define MT_PER_WAVE  2
#define MT_PER_BLOCK 4
#define MTILES_PER_E (T_TOK / 16)   // 1024
#define YBLK_PER_E  (MTILES_PER_E / MT_PER_BLOCK)  // 256

#define JT_PER_BLOCK 16                             // 16-col tiles per block
#define CHUNK_BYTES  (JT_PER_BLOCK * 32 * 16 * 2)   // 16 KB B slab per chunk
#define CHUNK_STRIDE ((size_t)64 * 32 * 16 * 2)     // bytes between chunks in Wb
#define NBUF 3                                      // triple buffer (48 KB LDS)

typedef __attribute__((ext_vector_type(16))) __bf16 v16bf;
typedef __attribute__((ext_vector_type(8)))  float  v8f;
typedef __attribute__((ext_vector_type(4)))  int    v4i;

#if __has_builtin(__builtin_amdgcn_global_load_async_to_lds_b128)
#define HAVE_ASYNC 1
#else
#define HAVE_ASYNC 0
#endif

#define GLOBAL_AS __attribute__((address_space(1)))
#define LDS_AS    __attribute__((address_space(3)))

__device__ __forceinline__ unsigned int f2bf_rne_u32(float f) {
    union { float f; unsigned int u; } v; v.f = f;
    unsigned int u = v.u;
    return (u + 0x7FFFu + ((u >> 16) & 1u)) >> 16;   // round-to-nearest-even
}

// ---------------------------------------------------------------------------
__global__ void init_counts(int* __restrict__ counts) {
    if (threadIdx.x < NEXP) counts[threadIdx.x] = 0;
}

__global__ void bin_tokens(const int* __restrict__ idxs,
                           int* __restrict__ counts,
                           int* __restrict__ perm) {
    int t = blockIdx.x * blockDim.x + threadIdx.x;
    if (t < T_TOK) {
        int e = idxs[t];
        int p = atomicAdd(&counts[e], 1);
        perm[e * T_TOK + p] = t;
    }
}

// x (fp32 [T][DIN]) -> xb (bf16 row-major). 4 floats per thread.
__global__ __launch_bounds__(256)
void convert_x(const float* __restrict__ x, unsigned short* __restrict__ xb) {
    int idx = blockIdx.x * 256 + threadIdx.x;
    float4 f = ((const float4*)x)[idx];
    uint2 o;
    o.x = f2bf_rne_u32(f.x) | (f2bf_rne_u32(f.y) << 16);
    o.y = f2bf_rne_u32(f.z) | (f2bf_rne_u32(f.w) << 16);
    ((uint2*)xb)[idx] = o;
}

// W[e][k][n] fp32 -> bf16 B fragments:
//   lane = (n&15) + 16*(k>=16); element t -> k = t + 16*(lane>=16)
// Wb element index: (((e*NCHUNKS + c)*64 + j)*32 + lane)*16 + t
// -> each lane's fragment is 32 contiguous bytes; a block's 16 j-tiles for one
//    chunk form one contiguous 16 KB slab (async-copy friendly).
__global__ __launch_bounds__(256)
void convert_w(const float* __restrict__ W, unsigned short* __restrict__ Wb) {
    int idx  = blockIdx.x * 256 + threadIdx.x;   // 524288 total
    int lane = idx & 31;
    int j    = (idx >> 5) & 63;
    int c    = (idx >> 11) & 31;
    int e    = idx >> 16;
    int n     = j * 16 + (lane & 15);
    int kbase = c * KCH + ((lane >> 4) ? 16 : 0);
    const float* src = W + ((size_t)e * DIN + kbase) * DOUT + n;
    union { unsigned short h[16]; uint4 q[2]; } o;
#pragma unroll
    for (int t = 0; t < 16; ++t)
        o.h[t] = (unsigned short)f2bf_rne_u32(src[(size_t)t * DOUT]);
    uint4* dst = (uint4*)(Wb + ((size_t)idx * 16));
    dst[0] = o.q[0];
    dst[1] = o.q[1];
}

// ---------------------------------------------------------------------------
__device__ __forceinline__ void issue_copy(const unsigned char* gsrc,
                                           unsigned char* ldst, int tid) {
#pragma unroll
    for (int i = 0; i < 4; ++i) {
        int off = tid * 16 + i * 4096;          // 256 thr * 16B * 4 = 16 KB
#if HAVE_ASYNC
        __builtin_amdgcn_global_load_async_to_lds_b128(
            (GLOBAL_AS v4i*)(gsrc + off),
            (LDS_AS v4i*)(ldst + off), 0, 0);
#else
        *(uint4*)(ldst + off) = *(const uint4*)(gsrc + off);
#endif
    }
}

__global__ __launch_bounds__(256)
void moe_gemm(const unsigned short* __restrict__ xb,
              const float* __restrict__ bias,
              const unsigned short* __restrict__ Wb,
              const int* __restrict__ counts,
              const int* __restrict__ perm,
              float* __restrict__ out) {
    __shared__ __align__(16) unsigned char ldsB[NBUF * CHUNK_BYTES];

    const int tid    = threadIdx.x;
    const int lane   = tid & 31;
    const int wid    = tid >> 5;
    const int wave_m = wid >> 2;             // 0..1
    const int wave_n = wid & 3;              // 0..3

    const int e     = blockIdx.y >> 8;       // 256 y-blocks per expert
    const int blkM0 = (blockIdx.y & (YBLK_PER_E - 1)) * MT_PER_BLOCK;
    const int cnt   = counts[e];
    if (blkM0 * 16 >= cnt) return;           // block-uniform (barriers below)

    const int mtile0 = blkM0 + wave_m * MT_PER_WAVE;
    const int n0  = blockIdx.x * BLOCK_N + wave_n * WAVE_N;
    const int cm1 = cnt - 1;
    const int* permE = perm + e * T_TOK;

    // --- A fragment rows (lanes 0-15 / 16-31 both map M = lane&15)
    const int lrow  = lane & 15;
    const int khalf = (lane >> 4) * 16;
    int g0 = mtile0 * 16 + lrow;          if (g0 > cm1) g0 = cm1;
    int g1 = (mtile0 + 1) * 16 + lrow;    if (g1 > cm1) g1 = cm1;
    const unsigned short* xrow0 = xb + (size_t)permE[g0] * DIN + khalf;
    const unsigned short* xrow1 = xb + (size_t)permE[g1] * DIN + khalf;

    // --- B slab base for this block (chunk 0): contiguous 16 KB per chunk
    const int jb0 = blockIdx.x * JT_PER_BLOCK;
    const unsigned char* slabE = (const unsigned char*)
        (Wb + (((size_t)e * NCHUNKS) * 64 + jb0) * 32 * 16);

    v8f acc[MT_PER_WAVE][4];
#pragma unroll
    for (int mt = 0; mt < MT_PER_WAVE; ++mt)
#pragma unroll
        for (int j = 0; j < 4; ++j)
            acc[mt][j] = (v8f){0.f,0.f,0.f,0.f,0.f,0.f,0.f,0.f};

    // prologue: stage chunks 0 and 1
    issue_copy(slabE,                ldsB,               tid);
    issue_copy(slabE + CHUNK_STRIDE, ldsB + CHUNK_BYTES, tid);

    const unsigned char* myB = ldsB + ((wave_n * 4) * 32 + lane) * 32;

    for (int c = 0; c < NCHUNKS; ++c) {
#if HAVE_ASYNC
        if (c == NCHUNKS - 1) asm volatile("s_wait_asynccnt 0x0" ::: "memory");
        else                  asm volatile("s_wait_asynccnt 0x4" ::: "memory");
#endif
        __syncthreads();   // chunk c resident everywhere; buf (c+2)%3 fully consumed
        if (c + 2 < NCHUNKS)
            issue_copy(slabE + (size_t)(c + 2) * CHUNK_STRIDE,
                       ldsB + ((c + 2) % NBUF) * CHUNK_BYTES, tid);

        // A: elements 0..15 = K(c*32+khalf+0..15)
        union { uint4 q[2]; v16bf v; } A0, A1;
        A0.q[0] = *(const uint4*)(xrow0 + c * KCH);
        A0.q[1] = *(const uint4*)(xrow0 + c * KCH + 8);
        A1.q[0] = *(const uint4*)(xrow1 + c * KCH);
        A1.q[1] = *(const uint4*)(xrow1 + c * KCH + 8);

        const unsigned char* bb = myB + (c % NBUF) * CHUNK_BYTES;
#pragma unroll
        for (int j = 0; j < 4; ++j) {
            union { uint4 q[2]; v16bf v; } B;
            B.q[0] = *(const uint4*)(bb + j * 1024);        // 32 lanes * 32 B
            B.q[1] = *(const uint4*)(bb + j * 1024 + 16);
            acc[0][j] = __builtin_amdgcn_wmma_f32_16x16x32_bf16(
                false, A0.v, false, B.v, (short)0, acc[0][j], false, false);
            acc[1][j] = __builtin_amdgcn_wmma_f32_16x16x32_bf16(
                false, A1.v, false, B.v, (short)0, acc[1][j], false, false);
        }
    }

    // --- epilogue: C layout lane n = lane&15, VGPR v -> m = v + 8*(lane>=16)
    const int nlane = lane & 15;
    const int mhi   = (lane >> 4) * 8;
    int  toks[MT_PER_WAVE][8];
    bool valid[MT_PER_WAVE][8];
#pragma unroll
    for (int mt = 0; mt < MT_PER_WAVE; ++mt) {
#pragma unroll
        for (int v = 0; v < 8; ++v) {
            int g = (mtile0 + mt) * 16 + mhi + v;
            valid[mt][v] = (g < cnt);
            toks[mt][v]  = permE[g > cm1 ? cm1 : g];   // clamped: branch-free load
        }
    }
#pragma unroll
    for (int j = 0; j < 4; ++j) {
        int n = n0 + j * 16 + nlane;
        float bv = bias[e * DOUT + n];
#pragma unroll
        for (int mt = 0; mt < MT_PER_WAVE; ++mt) {
#pragma unroll
            for (int v = 0; v < 8; ++v) {
                if (valid[mt][v]) {
                    float val = acc[mt][j][v] + bv;
                    out[(size_t)toks[mt][v] * DOUT + n] = val > 0.f ? val : 0.f;
                }
            }
        }
    }
}

// ---------------------------------------------------------------------------
extern "C" void kernel_launch(void* const* d_in, const int* in_sizes, int n_in,
                              void* d_out, int out_size, void* d_ws, size_t ws_size,
                              hipStream_t stream) {
    const float* x    = (const float*)d_in[0];   // [T, DIN]
    const int*   idxs = (const int*)  d_in[1];   // [T]
    const float* W    = (const float*)d_in[2];   // [E, DIN, DOUT]
    const float* b    = (const float*)d_in[3];   // [E, DOUT]
    float*       out  = (float*)d_out;           // [T, DOUT]

    char* ws = (char*)d_ws;
    int* counts = (int*)ws;                                    // pad to 256B
    int* perm   = (int*)(ws + 256);                            // 512 KB
    unsigned short* Wb =
        (unsigned short*)(ws + 256 + (size_t)NEXP * T_TOK * 4);          // 16 MB
    unsigned short* xbf =
        (unsigned short*)(ws + 256 + (size_t)NEXP * T_TOK * 4
                             + (size_t)NEXP * DIN * DOUT * 2);           // 32 MB

    init_counts<<<1, 32, 0, stream>>>(counts);
    bin_tokens<<<T_TOK / 256, 256, 0, stream>>>(idxs, counts, perm);
    convert_x<<<((size_t)T_TOK * DIN / 4) / 256, 256, 0, stream>>>(x, xbf);
    convert_w<<<(NEXP * NCHUNKS * 64 * 32) / 256, 256, 0, stream>>>(W, Wb);

    dim3 grid(DOUT / BLOCK_N, NEXP * YBLK_PER_E);  // (4, 2048)
    moe_gemm<<<grid, 256, 0, stream>>>(xbf, b, Wb, counts, perm, out);
}
